// GVPVectorField_27539330302128
// MI455X (gfx1250) — matrix operand order repaired and checked by reading, not compiled, not run on hardware
//
#include <hip/hip_runtime.h>
#include <math.h>

typedef __bf16 bf16;
typedef __attribute__((ext_vector_type(16))) __bf16 v16bf;
typedef __attribute__((ext_vector_type(8)))  float  v8f;

#define NNODES 20000
#define NEDGES 320000

__device__ __forceinline__ float siluf(float x){ return x / (1.f + __expf(-x)); }
__device__ __forceinline__ float sigmf(float x){ return 1.f / (1.f + __expf(-x)); }

// A-matrix (16x32 bf16 per k-chunk) ISA fragment-order swizzle:
// lane<16: row=lane, elems 0..7 -> K=0..7, 8..15 -> K=16..23
// lane>=16: row=lane-16, elems 0..7 -> K=8..15, 8..15 -> K=24..31
__device__ __forceinline__ int a_sw(int r, int k) {
  int kc = k >> 5, km = k & 31;
  int hh = (km >> 3) & 1;
  int e  = (km & 7) + ((km & 16) ? 8 : 0);
  return ((kc << 5) + (hh << 4) + r) * 16 + e;
}

// ---------------- weight prep: f32 [K][N] -> bf16 B-fragment order ----------------
__global__ __launch_bounds__(256)
void k_prep(const float* __restrict__ W, bf16* __restrict__ out,
            int K, int N, int Kpad, int Npad) {
  int i = blockIdx.x * 256 + threadIdx.x;
  int total = Kpad * Npad;
  if (i >= total) return;
  int k = i / Npad, n = i - k * Npad;
  float val = (k < K && n < N) ? W[k * N + n] : 0.f;
  int kc = k >> 5, km = k & 31;
  int hh = (km >> 3) & 1, e = (km & 7) + ((km & 16) ? 8 : 0);
  int nt = n >> 4, nl = n & 15;
  int NT = Npad >> 4;
  out[(((kc * NT + nt) << 5) + (hh << 4) + nl) * 16 + e] = (bf16)val;
}

// ---------------- generic GVP step on a 16-row tile (one wave) ----------------
template<int VIN, int HD, int KC, bool DOGATE>
__device__ __forceinline__ void gvp_step(int lane, bf16* A, float* Va, float* Vb, float* G,
    const float* Wh, const float* Wu, const float* bsv, const float* bgv,
    const bf16* __restrict__ Bws, const bf16* __restrict__ Bwg, int shOff, v8f* acc)
{
  // Vh = Va . Wh   -> Vb  (VALU)
  for (int t = lane; t < 16 * HD * 3; t += 32) {
    int r = t / (HD * 3); int rem = t - r * (HD * 3); int hh = rem / 3, c = rem - hh * 3;
    float s = 0.f;
    #pragma unroll
    for (int vv = 0; vv < VIN; ++vv) s += Va[r * 51 + vv * 3 + c] * Wh[vv * HD + hh];
    Vb[r * 51 + hh * 3 + c] = s;
  }
  __syncthreads();
  // sh = ||Vh||  -> A[:, shOff..]
  for (int t = lane; t < 16 * HD; t += 32) {
    int r = t / HD, hh = t - r * HD;
    float a0 = Vb[r*51+hh*3+0], a1 = Vb[r*51+hh*3+1], a2 = Vb[r*51+hh*3+2];
    A[a_sw(r, shOff + hh)] = (bf16)sqrtf(a0*a0 + a1*a1 + a2*a2 + 1e-12f);
  }
  __syncthreads();
  // feats_out = silu([feats, sh] @ Ws + bs)  -- WMMA bf16 -> f32
  v8f zero = {0.f,0.f,0.f,0.f,0.f,0.f,0.f,0.f};
  #pragma unroll
  for (int nt = 0; nt < 4; ++nt) acc[nt] = zero;
  #pragma unroll
  for (int kc = 0; kc < KC; ++kc) {
    v16bf af = *reinterpret_cast<const v16bf*>(A + ((kc << 5) + lane) * 16);
    #pragma unroll
    for (int nt = 0; nt < 4; ++nt) {
      v16bf bfr = *reinterpret_cast<const v16bf*>(Bws + (((kc * 4 + nt) << 5) + lane) * 16);
      acc[nt] = __builtin_amdgcn_wmma_f32_16x16x32_bf16(false, af, false, bfr, (short)0, acc[nt], false, false);
    }
  }
  int rb = (lane & 16) ? 8 : 0, ncl = lane & 15;
  #pragma unroll
  for (int nt = 0; nt < 4; ++nt) {
    float bias = bsv[nt * 16 + ncl];
    #pragma unroll
    for (int m = 0; m < 8; ++m) acc[nt][m] = siluf(acc[nt][m] + bias);
  }
  __syncthreads();
  #pragma unroll
  for (int nt = 0; nt < 4; ++nt)
    #pragma unroll
    for (int m = 0; m < 8; ++m)
      A[a_sw(rb + m, nt * 16 + ncl)] = (bf16)acc[nt][m];
  __syncthreads();
  if constexpr (DOGATE) {
    // gate = sigmoid(feats_out @ Wg + bg)  -- WMMA (K=64, N=16)
    v8f g = {0.f,0.f,0.f,0.f,0.f,0.f,0.f,0.f};
    #pragma unroll
    for (int kc = 0; kc < 2; ++kc) {
      v16bf af  = *reinterpret_cast<const v16bf*>(A + ((kc << 5) + lane) * 16);
      v16bf bfr = *reinterpret_cast<const v16bf*>(Bwg + ((kc << 5) + lane) * 16);
      g = __builtin_amdgcn_wmma_f32_16x16x32_bf16(false, af, false, bfr, (short)0, g, false, false);
    }
    float gb = bgv[ncl];
    #pragma unroll
    for (int m = 0; m < 8; ++m) G[(rb + m) * 16 + ncl] = sigmf(g[m] + gb);
    __syncthreads();
    // Va <- gate * (Vh @ Wu)
    for (int t = lane; t < 768; t += 32) {
      int r = t / 48, rem = t - r * 48; int u = rem / 3, c = rem - u * 3;
      float s = 0.f;
      #pragma unroll
      for (int hh = 0; hh < HD; ++hh) s += Vb[r * 51 + hh * 3 + c] * Wu[hh * 16 + u];
      Va[r * 51 + u * 3 + c] = G[r * 16 + u] * s;
    }
    __syncthreads();
  }
}

// ---------------- embeddings ----------------
__global__ __launch_bounds__(256)
void k_embed_node(const float* __restrict__ na, const float* __restrict__ nc,
                  const float* __restrict__ tt, const int* __restrict__ batch,
                  const float* W1, const float* b1, const float* W2, const float* b2,
                  const float* lg, const float* lb, float* __restrict__ hbuf)
{
  __shared__ float sW1[17*64], sW2[64*64], sb1[64], sb2[64], slg[64], slb[64];
  int tid = threadIdx.x;
  for (int i = tid; i < 17*64; i += 256) sW1[i] = W1[i];
  for (int i = tid; i < 64*64; i += 256) sW2[i] = W2[i];
  if (tid < 64) { sb1[tid]=b1[tid]; sb2[tid]=b2[tid]; slg[tid]=lg[tid]; slb[tid]=lb[tid]; }
  __syncthreads();
  int n = blockIdx.x * 256 + tid;
  if (n >= NNODES) return;
  float in[17];
  #pragma unroll
  for (int j = 0; j < 10; ++j) in[j] = na[(size_t)n*10 + j];
  #pragma unroll
  for (int j = 0; j < 6; ++j) in[10+j] = nc[(size_t)n*6 + j];
  in[16] = tt[batch[n]];
  float h1[64];
  for (int o = 0; o < 64; ++o) {
    float s = sb1[o];
    #pragma unroll
    for (int j = 0; j < 17; ++j) s += in[j] * sW1[j*64 + o];
    h1[o] = siluf(s);
  }
  float h2[64]; float mu = 0.f;
  for (int o = 0; o < 64; ++o) {
    float s = sb2[o];
    for (int j = 0; j < 64; ++j) s += h1[j] * sW2[j*64 + o];
    s = siluf(s); h2[o] = s; mu += s;
  }
  mu *= (1.f/64.f);
  float var = 0.f;
  for (int o = 0; o < 64; ++o) { float d = h2[o]-mu; var += d*d; }
  float inv = rsqrtf(var*(1.f/64.f) + 1e-5f);
  for (int o = 0; o < 64; ++o) hbuf[(size_t)n*64+o] = (h2[o]-mu)*inv*slg[o] + slb[o];
}

__global__ __launch_bounds__(256)
void k_embed_edge(const float* __restrict__ et,
                  const float* W1, const float* b1, const float* W2, const float* b2,
                  const float* lg, const float* lb, float* __restrict__ ebuf)
{
  __shared__ float sW1[5*64], sW2[64*64], sb1[64], sb2[64], slg[64], slb[64];
  int tid = threadIdx.x;
  for (int i = tid; i < 5*64; i += 256) sW1[i] = W1[i];
  for (int i = tid; i < 64*64; i += 256) sW2[i] = W2[i];
  if (tid < 64) { sb1[tid]=b1[tid]; sb2[tid]=b2[tid]; slg[tid]=lg[tid]; slb[tid]=lb[tid]; }
  __syncthreads();
  int n = blockIdx.x * 256 + tid;
  if (n >= NEDGES) return;
  float in[5];
  #pragma unroll
  for (int j = 0; j < 5; ++j) in[j] = et[(size_t)n*5 + j];
  float h1[64];
  for (int o = 0; o < 64; ++o) {
    float s = sb1[o];
    #pragma unroll
    for (int j = 0; j < 5; ++j) s += in[j] * sW1[j*64 + o];
    h1[o] = siluf(s);
  }
  float h2[64]; float mu = 0.f;
  for (int o = 0; o < 64; ++o) {
    float s = sb2[o];
    for (int j = 0; j < 64; ++j) s += h1[j] * sW2[j*64 + o];
    s = siluf(s); h2[o] = s; mu += s;
  }
  mu *= (1.f/64.f);
  float var = 0.f;
  for (int o = 0; o < 64; ++o) { float d = h2[o]-mu; var += d*d; }
  float inv = rsqrtf(var*(1.f/64.f) + 1e-5f);
  for (int o = 0; o < 64; ++o) ebuf[(size_t)n*64+o] = (h2[o]-mu)*inv*slg[o] + slb[o];
}

// ---------------- per-edge message kernel: 4 waves x 16 edges ----------------
__global__ __launch_bounds__(128)
void k_message(const float* __restrict__ hb, const float* __restrict__ vbuf,
               const float* __restrict__ xb, const float* __restrict__ ebuf,
               float* __restrict__ smb, float* __restrict__ vmb,
               const int* __restrict__ src, const int* __restrict__ dst,
               const bf16* __restrict__ wbase,
               const float* Wh0, const float* Wu0, const float* bs0, const float* bg0,
               const float* Wh1, const float* Wu1, const float* bs1, const float* bg1,
               const float* Wh2, const float* Wu2, const float* bs2, const float* bg2)
{
  __shared__ alignas(32) bf16 sA[4][192*16];
  __shared__ float sVa[4][16*51];
  __shared__ float sVb[4][16*51];
  __shared__ float sG[4][256];
  __shared__ int sS[4][16], sD[4][16];
  __shared__ float cWh0[17*17], cWu0[17*16];
  __shared__ float cWh1[256], cWu1[256], cWh2[256], cWu2[256];
  __shared__ float cbs[3][64], cbg[3][16];
  int tid = threadIdx.x, w = tid >> 5, lane = tid & 31;
  for (int i = tid; i < 289; i += 128) cWh0[i] = Wh0[i];
  for (int i = tid; i < 272; i += 128) cWu0[i] = Wu0[i];
  for (int i = tid; i < 256; i += 128) { cWh1[i]=Wh1[i]; cWu1[i]=Wu1[i]; cWh2[i]=Wh2[i]; cWu2[i]=Wu2[i]; }
  if (tid < 64) { cbs[0][tid]=bs0[tid]; cbs[1][tid]=bs1[tid]; cbs[2][tid]=bs2[tid]; }
  if (tid < 16) { cbg[0][tid]=bg0[tid]; cbg[1][tid]=bg1[tid]; cbg[2][tid]=bg2[tid]; }
  int e0 = (blockIdx.x * 4 + w) * 16;
  if (lane < 16) {
    int e = e0 + lane;
    int s = src[e], d = dst[e];
    sS[w][lane] = s; sD[w][lane] = d;
    float x0 = xb[s*3+0]-xb[d*3+0];
    float x1 = xb[s*3+1]-xb[d*3+1];
    float x2 = xb[s*3+2]-xb[d*3+2];
    float dij = sqrtf(x0*x0+x1*x1+x2*x2 + 1e-12f);
    float inv = 1.f/dij;
    sVa[w][lane*51+0] = x0*inv; sVa[w][lane*51+1] = x1*inv; sVa[w][lane*51+2] = x2*inv;
    #pragma unroll
    for (int j = 0; j < 16; ++j) {       // rbf: mu=linspace(0,20,16), sigma=1.25
      float u = (dij - (20.f/15.f)*(float)j) * 0.8f;
      sA[w][a_sw(lane, 64+j)] = (bf16)__expf(-u*u);
    }
  }
  __syncthreads();
  for (int t = lane; t < 1024; t += 32) {
    int r = t >> 6, k = t & 63;
    sA[w][a_sw(r, k)]      = (bf16)hb[(size_t)sS[w][r]*64 + k];
    sA[w][a_sw(r, 80 + k)] = (bf16)ebuf[(size_t)(e0+r)*64 + k];
  }
  for (int t = lane; t < 768; t += 32) {
    int r = t / 48, rem = t - r*48; int u = rem/3, c = rem - u*3;
    sVa[w][r*51 + (1+u)*3 + c] = vbuf[(size_t)sS[w][r]*48 + rem];
  }
  for (int t = lane; t < 16*31; t += 32) {  // pad cols 161..191
    int r = t / 31, k = 161 + (t - r*31);
    sA[w][a_sw(r, k)] = (bf16)0.f;
  }
  __syncthreads();
  v8f acc[4];
  gvp_step<17,17,6,true>(lane, sA[w], sVa[w], sVb[w], sG[w], cWh0, cWu0, cbs[0], cbg[0],
                         wbase + 0, wbase + 24576, 144, acc);
  for (int t = lane; t < 256; t += 32) sA[w][a_sw(t>>4, 80+(t&15))] = (bf16)0.f;
  __syncthreads();
  gvp_step<16,16,3,true>(lane, sA[w], sVa[w], sVb[w], sG[w], cWh1, cWu1, cbs[1], cbg[1],
                         wbase + 12288, wbase + 25600, 64, acc);
  gvp_step<16,16,3,true>(lane, sA[w], sVa[w], sVb[w], sG[w], cWh2, cWu2, cbs[2], cbg[2],
                         wbase + 18432, wbase + 26624, 64, acc);
  // scatter: sm += sca/100 (from D fragments), vm += vec/100
  int rb = (lane & 16) ? 8 : 0, ncl = lane & 15;
  #pragma unroll
  for (int nt = 0; nt < 4; ++nt)
    #pragma unroll
    for (int m = 0; m < 8; ++m)
      atomicAdd(&smb[(size_t)sD[w][rb+m]*64 + nt*16 + ncl], acc[nt][m] * 0.01f);
  for (int t = lane; t < 768; t += 32) {
    int r = t / 48, rem = t - r*48; int u = rem/3, c = rem - u*3;
    atomicAdd(&vmb[(size_t)sD[w][r]*48 + rem], sVa[w][r*51 + u*3 + c] * 0.01f);
  }
}

// ---------------- node update phase A: LN(h+sm), v+=vm, zero sm/vm ----------------
__global__ __launch_bounds__(256)
void k_update_a(float* __restrict__ hb, float* __restrict__ vbuf,
                float* __restrict__ smb, float* __restrict__ vmb,
                const float* __restrict__ lg, const float* __restrict__ lb)
{
  int n = blockIdx.x * 256 + threadIdx.x;
  if (n >= NNODES) return;
  float buf[64]; float mu = 0.f;
  for (int k = 0; k < 64; ++k) {
    float x = hb[(size_t)n*64+k] + smb[(size_t)n*64+k];
    smb[(size_t)n*64+k] = 0.f;
    buf[k] = x; mu += x;
  }
  mu *= (1.f/64.f);
  float var = 0.f;
  for (int k = 0; k < 64; ++k) { float d = buf[k]-mu; var += d*d; }
  float inv = rsqrtf(var*(1.f/64.f) + 1e-5f);
  for (int k = 0; k < 64; ++k) hb[(size_t)n*64+k] = (buf[k]-mu)*inv*lg[k] + lb[k];
  for (int j = 0; j < 48; ++j) { vbuf[(size_t)n*48+j] += vmb[(size_t)n*48+j]; vmb[(size_t)n*48+j] = 0.f; }
}

// ---------------- node update phase B: 3 update GVPs + residual LN ----------------
__global__ __launch_bounds__(64)
void k_update_b(float* __restrict__ hb, float* __restrict__ vbuf,
                const bf16* __restrict__ wbase,
                const float* Wh0, const float* Wu0, const float* bs0, const float* bg0,
                const float* Wh1, const float* Wu1, const float* bs1, const float* bg1,
                const float* Wh2, const float* Wu2, const float* bs2, const float* bg2,
                const float* lng, const float* lnb)
{
  __shared__ alignas(32) bf16 sA[2][96*16];
  __shared__ float sVa[2][16*51], sVb[2][16*51], sG[2][256];
  __shared__ float sHm[2][16*64], sVc[2][16*48];
  __shared__ float cWh[3][256], cWu[3][256], cbs[3][64], cbg[3][16];
  int tid = threadIdx.x, w = tid >> 5, lane = tid & 31;
  for (int i = tid; i < 256; i += 64) {
    cWh[0][i]=Wh0[i]; cWh[1][i]=Wh1[i]; cWh[2][i]=Wh2[i];
    cWu[0][i]=Wu0[i]; cWu[1][i]=Wu1[i]; cWu[2][i]=Wu2[i];
  }
  if (tid < 64) { cbs[0][tid]=bs0[tid]; cbs[1][tid]=bs1[tid]; cbs[2][tid]=bs2[tid]; }
  if (tid < 16) { cbg[0][tid]=bg0[tid]; cbg[1][tid]=bg1[tid]; cbg[2][tid]=bg2[tid]; }
  int n0 = (blockIdx.x * 2 + w) * 16;
  for (int t = lane; t < 1024; t += 32) {
    int r = t >> 6, k = t & 63;
    float hv = hb[(size_t)(n0+r)*64 + k];
    sHm[w][r*64+k] = hv;
    sA[w][a_sw(r, k)] = (bf16)hv;
  }
  for (int t = lane; t < 768; t += 32) {
    int r = t / 48, rem = t - r*48; int u = rem/3, c = rem - u*3;
    float vv = vbuf[(size_t)(n0+r)*48 + rem];
    sVc[w][t] = vv;
    sVa[w][r*51 + u*3 + c] = vv;
  }
  for (int t = lane; t < 256; t += 32) sA[w][a_sw(t>>4, 80+(t&15))] = (bf16)0.f;
  __syncthreads();
  v8f acc[4];
  for (int j = 0; j < 3; ++j)
    gvp_step<16,16,3,true>(lane, sA[w], sVa[w], sVb[w], sG[w],
        cWh[j], cWu[j], cbs[j], cbg[j],
        wbase + j*6144, wbase + 18432 + j*1024, 64, acc);
  int rb = (lane & 16) ? 8 : 0, ncl = lane & 15;
  #pragma unroll
  for (int nt = 0; nt < 4; ++nt)
    #pragma unroll
    for (int m = 0; m < 8; ++m)
      sHm[w][(rb+m)*64 + nt*16 + ncl] += acc[nt][m];
  __syncthreads();
  if (lane < 16) {
    int r = lane; float mu = 0.f;
    for (int k = 0; k < 64; ++k) mu += sHm[w][r*64+k];
    mu *= (1.f/64.f);
    float var = 0.f;
    for (int k = 0; k < 64; ++k) { float d = sHm[w][r*64+k]-mu; var += d*d; }
    float inv = rsqrtf(var*(1.f/64.f) + 1e-5f);
    for (int k = 0; k < 64; ++k)
      hb[(size_t)(n0+r)*64+k] = (sHm[w][r*64+k]-mu)*inv*lng[k] + lnb[k];
  }
  for (int t = lane; t < 768; t += 32) {
    int r = t / 48, rem = t - r*48; int u = rem/3, c = rem - u*3;
    vbuf[(size_t)(n0+r)*48 + rem] = sVc[w][t] + sVa[w][r*51 + u*3 + c];
  }
}

// ---------------- position update: 3 GVPs, last vout=1, identity gate ----------------
__global__ __launch_bounds__(64)
void k_pos(const float* __restrict__ hb, const float* __restrict__ vbuf, float* __restrict__ xb,
           const bf16* __restrict__ wbase,
           const float* Wh0, const float* Wu0, const float* bs0, const float* bg0,
           const float* Wh1, const float* Wu1, const float* bs1, const float* bg1,
           const float* Wh2, const float* Wu2p, const float* bs2, const float* Wg2, const float* bg2)
{
  __shared__ alignas(32) bf16 sA[2][96*16];
  __shared__ float sVa[2][16*51], sVb[2][16*51], sG[2][256];
  __shared__ float cWh[3][256], cWu[2][256], cWu2[16], cbs[3][64], cbg[2][16], cWg2[64];
  __shared__ float cbg2;
  int tid = threadIdx.x, w = tid >> 5, lane = tid & 31;
  for (int i = tid; i < 256; i += 64) {
    cWh[0][i]=Wh0[i]; cWh[1][i]=Wh1[i]; cWh[2][i]=Wh2[i];
    cWu[0][i]=Wu0[i]; cWu[1][i]=Wu1[i];
  }
  if (tid < 64) { cbs[0][tid]=bs0[tid]; cbs[1][tid]=bs1[tid]; cbs[2][tid]=bs2[tid]; cWg2[tid]=Wg2[tid]; }
  if (tid < 16) { cbg[0][tid]=bg0[tid]; cbg[1][tid]=bg1[tid]; cWu2[tid]=Wu2p[tid]; }
  if (tid == 0) cbg2 = bg2[0];
  int n0 = (blockIdx.x * 2 + w) * 16;
  for (int t = lane; t < 1024; t += 32) {
    int r = t >> 6, k = t & 63;
    sA[w][a_sw(r, k)] = (bf16)hb[(size_t)(n0+r)*64 + k];
  }
  for (int t = lane; t < 768; t += 32) {
    int r = t / 48, rem = t - r*48; int u = rem/3, c = rem - u*3;
    sVa[w][r*51 + u*3 + c] = vbuf[(size_t)(n0+r)*48 + rem];
  }
  for (int t = lane; t < 256; t += 32) sA[w][a_sw(t>>4, 80+(t&15))] = (bf16)0.f;
  __syncthreads();
  v8f acc[4];
  gvp_step<16,16,3,true>(lane, sA[w], sVa[w], sVb[w], sG[w], cWh[0], cWu[0], cbs[0], cbg[0],
                         wbase + 0,    wbase + 18432, 64, acc);
  gvp_step<16,16,3,true>(lane, sA[w], sVa[w], sVb[w], sG[w], cWh[1], cWu[1], cbs[1], cbg[1],
                         wbase + 6144, wbase + 19456, 64, acc);
  gvp_step<16,16,3,false>(lane, sA[w], sVa[w], sVb[w], sG[w], cWh[2], cWu2, cbs[2], cbg[0],
                          wbase + 12288, wbase + 18432, 64, acc);
  if (lane < 16) {
    int r = lane;
    float g = cbg2;
    for (int o = 0; o < 64; ++o) g += (float)sA[w][a_sw(r, o)] * cWg2[o];
    #pragma unroll
    for (int c = 0; c < 3; ++c) {
      float s = 0.f;
      #pragma unroll
      for (int hh = 0; hh < 16; ++hh) s += sVb[w][r*51 + hh*3 + c] * cWu2[hh];
      xb[(size_t)(n0+r)*3 + c] += g * s;
    }
  }
}

// ---------------- host ----------------
extern "C" void kernel_launch(void* const* d_in, const int* in_sizes, int n_in,
                              void* d_out, int out_size, void* d_ws, size_t ws_size,
                              hipStream_t stream) {
  (void)in_sizes; (void)n_in; (void)out_size; (void)ws_size;
  const float* node_a = (const float*)d_in[0];
  const float* node_c = (const float*)d_in[1];
  const float* x_t    = (const float*)d_in[2];
  const float* e_t    = (const float*)d_in[3];
  const float* tsc    = (const float*)d_in[4];
  const int*   batch  = (const int*)d_in[5];
  const int*   src    = (const int*)d_in[6];
  const int*   dst    = (const int*)d_in[7];
  auto Pf = [&](int i){ return (const float*)d_in[i]; };

  bf16*  wbf = (bf16*)d_ws;
  float* F   = (float*)((char*)d_ws + 434176);   // after 217088 bf16 weights
  float* hb  = F;
  float* vb  = hb + 1280000;
  float* xb  = vb + 960000;
  float* eb  = xb + 60000;
  float* smb = eb + 20480000;
  float* vmb = smb + 1280000;

  auto prep = [&](const float* W, size_t off, int K, int N, int Kpad, int Npad) {
    int total = Kpad * Npad;
    k_prep<<<(total + 255) / 256, 256, 0, stream>>>(W, wbf + off, K, N, Kpad, Npad);
  };
  // per-conv bf16 layout: Ws0@0(12288) Ws1@12288 Ws2@18432 Wg0@24576 Wg1@25600 Wg2@26624
  //                       updWs[j]@27648+j*6144  updWg[j]@46080+j*1024   (stride 49152)
  for (int ci = 0; ci < 4; ++ci) {
    size_t base = (size_t)ci * 49152;
    int cb = 20 + ci * 40;
    prep(Pf(cb + 0*6 + 2), base + 0,     161, 64, 192, 64);
    prep(Pf(cb + 1*6 + 2), base + 12288,  80, 64,  96, 64);
    prep(Pf(cb + 2*6 + 2), base + 18432,  80, 64,  96, 64);
    prep(Pf(cb + 0*6 + 4), base + 24576,  64, 16,  64, 16);
    prep(Pf(cb + 1*6 + 4), base + 25600,  64, 16,  64, 16);
    prep(Pf(cb + 2*6 + 4), base + 26624,  64, 16,  64, 16);
    for (int j = 0; j < 3; ++j) {
      prep(Pf(cb + 18 + j*6 + 2), base + 27648 + (size_t)j*6144, 80, 64, 96, 64);
      prep(Pf(cb + 18 + j*6 + 4), base + 46080 + (size_t)j*1024, 64, 16, 64, 16);
    }
  }
  const size_t pb = 196608;   // pos_upd weights
  for (int j = 0; j < 3; ++j) prep(Pf(180 + j*6 + 2), pb + (size_t)j*6144, 80, 64, 96, 64);
  prep(Pf(180 + 0*6 + 4), pb + 18432, 64, 16, 64, 16);
  prep(Pf(180 + 1*6 + 4), pb + 19456, 64, 16, 64, 16);

  hipMemsetAsync(smb, 0, 1280000 * sizeof(float), stream);
  hipMemsetAsync(vmb, 0,  960000 * sizeof(float), stream);
  hipMemsetAsync(vb,  0,  960000 * sizeof(float), stream);
  hipMemcpyAsync(xb, x_t, 60000 * sizeof(float), hipMemcpyDeviceToDevice, stream);

  k_embed_node<<<(NNODES + 255) / 256, 256, 0, stream>>>(node_a, node_c, tsc, batch,
      Pf(8), Pf(9), Pf(10), Pf(11), Pf(12), Pf(13), hb);
  k_embed_edge<<<(NEDGES + 255) / 256, 256, 0, stream>>>(e_t,
      Pf(14), Pf(15), Pf(16), Pf(17), Pf(18), Pf(19), eb);

  for (int ci = 0; ci < 4; ++ci) {
    int cb = 20 + ci * 40;
    size_t base = (size_t)ci * 49152;
    k_message<<<NEDGES / 64, 128, 0, stream>>>(hb, vb, xb, eb, smb, vmb, src, dst,
        wbf + base,
        Pf(cb + 0),  Pf(cb + 1),  Pf(cb + 3),  Pf(cb + 5),
        Pf(cb + 6),  Pf(cb + 7),  Pf(cb + 9),  Pf(cb + 11),
        Pf(cb + 12), Pf(cb + 13), Pf(cb + 15), Pf(cb + 17));
    k_update_a<<<(NNODES + 255) / 256, 256, 0, stream>>>(hb, vb, smb, vmb, Pf(cb + 36), Pf(cb + 37));
    k_update_b<<<NNODES / 32, 64, 0, stream>>>(hb, vb, wbf + base + 27648,
        Pf(cb + 18), Pf(cb + 19), Pf(cb + 21), Pf(cb + 23),
        Pf(cb + 24), Pf(cb + 25), Pf(cb + 27), Pf(cb + 29),
        Pf(cb + 30), Pf(cb + 31), Pf(cb + 33), Pf(cb + 35),
        Pf(cb + 38), Pf(cb + 39));
    if (ci == 1 || ci == 3) {
      k_pos<<<NNODES / 32, 64, 0, stream>>>(hb, vb, xb, wbf + pb,
          Pf(180 + 0),  Pf(180 + 1),  Pf(180 + 3),  Pf(180 + 5),
          Pf(180 + 6),  Pf(180 + 7),  Pf(180 + 9),  Pf(180 + 11),
          Pf(180 + 12), Pf(180 + 13), Pf(180 + 15), Pf(180 + 16), Pf(180 + 17));
    }
  }

  float* out = (float*)d_out;
  hipMemcpyAsync(out,           hb, 1280000 * sizeof(float), hipMemcpyDeviceToDevice, stream);
  hipMemcpyAsync(out + 1280000, xb,   60000 * sizeof(float), hipMemcpyDeviceToDevice, stream);
  hipMemcpyAsync(out + 1340000, vb,  960000 * sizeof(float), hipMemcpyDeviceToDevice, stream);
}
